// Multi_Head_Self_Attention_85126251807373
// MI455X (gfx1250) — matrix operand order repaired and checked by reading, not compiled, not run on hardware
//
#include <hip/hip_runtime.h>
#include <hip/hip_bf16.h>

// MHA forward: B=32, S=512, D=1024, H=16, DK=64.
// d_out = context [B,S,D] f32 (16.7M) ++ attn [B,H,S,S] f32 (134M).
// d_ws (__bf16 regions):
//   qbf  [0,        16777216)   : Q converted to bf16          (32MB)
//   wbf  [16777216, 19922944)   : Wq|Wk|Wv bf16                ( 6MB)
//   q_ws next 16777216          : q proj  [BH,S,DK] bf16
//   k_ws next 16777216          : k proj  [BH,S,DK] bf16
//   v_ws next 16777216          : v proj  [BH,DK,S] bf16 (transposed)

#define Bc 32
#define Sc 512
#define Dc 1024
#define Hc 16
#define DKc 64

#define QBF_OFF   0
#define WBF_OFF   16777216
#define QWS_OFF   19922944
#define KWS_OFF   (QWS_OFF + 16777216)
#define VWS_OFF   (KWS_OFF + 16777216)

typedef __attribute__((ext_vector_type(16))) __bf16 v16bf;
typedef __attribute__((ext_vector_type(8)))  float  v8f;

// 16 contiguous bf16 (32B, aligned) -> B-operand layout element order
static __device__ inline v16bf load16_contig(const __bf16* p) {
    union { uint4 u[2]; v16bf v; } t;
    t.u[0] = *(const uint4*)p;
    t.u[1] = *(const uint4*)(p + 8);
    return t.v;
}
// A-operand: elems 0..15 = p[0..15] (two 16B runs)
static __device__ inline v16bf load_a_split(const __bf16* p) {
    union { uint4 u[2]; v16bf v; } t;
    t.u[0] = *(const uint4*)p;
    t.u[1] = *(const uint4*)(p + 8);
    return t.v;
}

// ---------------------------------------------------------------------------
// Kernel 0: f32 -> bf16 bulk convert (n multiple of 8), packed b128 stores.
// ---------------------------------------------------------------------------
__global__ __launch_bounds__(256) void cvt_bf16_kernel(
    const float* __restrict__ src, __bf16* __restrict__ dst, int n)
{
    const int i = (blockIdx.x * 256 + threadIdx.x) * 8;
    if (i + 8 > n) return;
    union { __bf16 h[8]; uint4 u; } t;
    #pragma unroll
    for (int j = 0; j < 8; ++j) t.h[j] = (__bf16)src[i + j];
    *(uint4*)(dst + i) = t.u;
}

// ---------------------------------------------------------------------------
// Kernel 1: QKV projection, all-bf16 operands.
// One wave: 16 rows x 64 cols (4 accumulators), K-loop of 32.
// Jobs: mat(3) x mt(1024) x ngroup(16) = 49152 waves -> 6144 blocks.
// ---------------------------------------------------------------------------
__global__ __launch_bounds__(256) void qkv_proj_kernel(
    const __bf16* __restrict__ wsb,
    const float* __restrict__ bq, const float* __restrict__ bk,
    const float* __restrict__ bv,
    __bf16* __restrict__ wso)
{
    const int lane = threadIdx.x & 31;
    const int wave = threadIdx.x >> 5;
    const int job  = blockIdx.x * 8 + wave;

    const int mat = job / (1024 * 16);           // 0=q 1=k 2=v
    const int rem = job % (1024 * 16);
    const int m0  = (rem >> 4) * 16;
    const int n0  = (rem & 15) * 64;

    const float* bias = (mat == 0) ? bq : (mat == 1) ? bk : bv;
    const __bf16* A   = wsb + QBF_OFF;
    const __bf16* W   = wsb + WBF_OFF + (size_t)mat * (Dc * Dc);

    const int lm = lane & 15;
    const int lh = lane >> 4;

    const __bf16* arow = A + (size_t)(m0 + lm) * Dc;
    const __bf16* br0  = W + (size_t)(n0 +  0 + lm) * Dc;
    const __bf16* br1  = W + (size_t)(n0 + 16 + lm) * Dc;
    const __bf16* br2  = W + (size_t)(n0 + 32 + lm) * Dc;
    const __bf16* br3  = W + (size_t)(n0 + 48 + lm) * Dc;

    v8f c0 = {}, c1 = {}, c2 = {}, c3 = {};
    for (int kk = 0; kk < Dc; kk += 32) {
        const v16bf a  = load_a_split(arow + kk + lh * 16);
        const v16bf b0 = load16_contig(br0 + kk + lh * 16);
        const v16bf b1 = load16_contig(br1 + kk + lh * 16);
        const v16bf b2 = load16_contig(br2 + kk + lh * 16);
        const v16bf b3 = load16_contig(br3 + kk + lh * 16);
        c0 = __builtin_amdgcn_wmma_f32_16x16x32_bf16(false, a, false, b0, (short)0, c0, false, false);
        c1 = __builtin_amdgcn_wmma_f32_16x16x32_bf16(false, a, false, b1, (short)0, c1, false, false);
        c2 = __builtin_amdgcn_wmma_f32_16x16x32_bf16(false, a, false, b2, (short)0, c2, false, false);
        c3 = __builtin_amdgcn_wmma_f32_16x16x32_bf16(false, a, false, b3, (short)0, c3, false, false);
    }

    __bf16* q_ws = wso + QWS_OFF;
    __bf16* k_ws = wso + KWS_OFF;
    __bf16* v_ws = wso + VWS_OFF;

    #pragma unroll
    for (int j = 0; j < 4; ++j) {
        const v8f* cj = (j == 0) ? &c0 : (j == 1) ? &c1 : (j == 2) ? &c2 : &c3;
        const int  n  = n0 + j * 16 + lm;
        const float bval = bias[n];
        const int h = n >> 6, dk = n & 63;
        #pragma unroll
        for (int r = 0; r < 8; ++r) {
            const int m = m0 + r + lh * 8;
            const int b = m >> 9, s = m & 511;
            const int bh = b * Hc + h;
            const __bf16 val = (__bf16)((*cj)[r] + bval);
            if (mat == 0)       q_ws[((size_t)bh * Sc + s) * DKc + dk] = val;
            else if (mat == 1)  k_ws[((size_t)bh * Sc + s) * DKc + dk] = val;
            else                v_ws[((size_t)bh * DKc + dk) * Sc + s] = val;   // transposed
        }
    }
}

// ---------------------------------------------------------------------------
// Kernel 2: attention, one (b,h, 32-query block) per 256-thread workgroup.
// LDS: exp-scores tile [32][512] f32 = 64KB exactly.
// ---------------------------------------------------------------------------
__global__ __launch_bounds__(256) void attn_kernel(
    const __bf16* __restrict__ wsb,
    const int*    __restrict__ length,
    float* __restrict__ ctx_out,     // [B,S,H*DK]
    float* __restrict__ attn_out)    // [B,H,S,S]
{
    __shared__ float sc[32 * 512];   // 64KB

    const int tid  = threadIdx.x;
    const int lane = tid & 31;
    const int wave = tid >> 5;
    const int lm   = lane & 15;
    const int lh   = lane >> 4;

    const int qblk = blockIdx.x & 15;        // S/32
    const int bh   = blockIdx.x >> 4;        // 0..511
    const int b    = bh >> 4;
    const int h    = bh & 15;
    const int q0   = qblk * 32;
    const int len  = length[b];

    const __bf16* qm = wsb + QWS_OFF + (size_t)bh * Sc * DKc;   // [S,DK]
    const __bf16* km = wsb + KWS_OFF + (size_t)bh * Sc * DKc;   // [S,DK]
    const __bf16* vt = wsb + VWS_OFF + (size_t)bh * DKc * Sc;   // [DK,S]

    const int qsub = wave >> 2;              // 0..1

    // ---- phase 1: scores = exp(q k^T / 8) * mask -> LDS ----
    const __bf16* qp = qm + (size_t)(q0 + qsub * 16 + lm) * DKc;
    const v16bf a0 = load_a_split(qp + lh * 16);        // K 0..31
    const v16bf a1 = load_a_split(qp + 32 + lh * 16);   // K 32..63

    const int kq = wave & 3;
    for (int kt = 0; kt < 8; ++kt) {
        const int key = kq * 128 + kt * 16 + lm;
        const __bf16* kp = km + (size_t)key * DKc;
        const v16bf b0 = load16_contig(kp + lh * 16);        // dk 0..31
        const v16bf b1 = load16_contig(kp + 32 + lh * 16);   // dk 32..63
        v8f c = {};
        c = __builtin_amdgcn_wmma_f32_16x16x32_bf16(false, a0, false, b0, (short)0, c, false, false);
        c = __builtin_amdgcn_wmma_f32_16x16x32_bf16(false, a1, false, b1, (short)0, c, false, false);
        const float msk = (key < len) ? 1.0f : 0.0f;
        #pragma unroll
        for (int r = 0; r < 8; ++r) {
            const int row = qsub * 16 + r + lh * 8;
            sc[row * 512 + key] = __expf(c[r] * 0.125f) * msk;   // raw exp (faithful)
        }
    }
    __syncthreads();

    // ---- phase 2+3: row sums (8 lanes/row, shfl), normalize, emit attn ----
    {
        const int row = tid >> 3;            // 0..31
        const int sub = tid & 7;
        float* p = sc + row * 512 + sub * 64;
        float s = 0.0f;
        for (int i = 0; i < 64; ++i) s += p[i];
        s += __shfl_xor(s, 4, 8);
        s += __shfl_xor(s, 2, 8);
        s += __shfl_xor(s, 1, 8);
        const float inv = 1.0f / (s + 1e-8f);
        float* arow = attn_out + ((size_t)bh * Sc + q0 + row) * Sc + sub * 64;
        for (int i = 0; i < 64; ++i) {
            const float v = p[i] * inv;
            p[i] = v;
            arow[i] = v;
        }
    }
    __syncthreads();

    // ---- phase 4: context = attn @ V  (V transposed -> contiguous B loads) ----
    {
        const int dt = wave & 3;             // dk tile
        const __bf16* vrow = vt + (size_t)(dt * 16 + lm) * Sc;  // column n = dk
        v8f c = {};
        for (int kk = 0; kk < Sc; kk += 32) {
            v16bf a;
            const float* ap = sc + (qsub * 16 + lm) * 512 + kk + lh * 16;
            #pragma unroll
            for (int e = 0; e < 8; ++e) { a[e] = (__bf16)ap[e]; a[8 + e] = (__bf16)ap[8 + e]; }
            const v16bf bm = load16_contig(vrow + kk + lh * 16);
            c = __builtin_amdgcn_wmma_f32_16x16x32_bf16(false, a, false, bm, (short)0, c, false, false);
        }
        #pragma unroll
        for (int r = 0; r < 8; ++r) {
            const int q  = q0 + qsub * 16 + r + lh * 8;
            const int dk = dt * 16 + lm;
            ctx_out[(((size_t)(b * Sc + q)) * Hc + h) * DKc + dk] = c[r];
        }
    }
}

// ---------------------------------------------------------------------------
extern "C" void kernel_launch(void* const* d_in, const int* in_sizes, int n_in,
                              void* d_out, int out_size, void* d_ws, size_t ws_size,
                              hipStream_t stream) {
    const float* Q      = (const float*)d_in[0];
    const float* Wq     = (const float*)d_in[1];
    const float* bq     = (const float*)d_in[2];
    const float* Wk     = (const float*)d_in[3];
    const float* bk     = (const float*)d_in[4];
    const float* Wv     = (const float*)d_in[5];
    const float* bv     = (const float*)d_in[6];
    const int*   length = (const int*)d_in[7];

    float*  ctx  = (float*)d_out;
    float*  attn = (float*)d_out + (size_t)Bc * Sc * Dc;
    __bf16* ws   = (__bf16*)d_ws;

    // 0) convert Q and weights to bf16 once
    cvt_bf16_kernel<<<(Bc * Sc * Dc) / 8 / 256, 256, 0, stream>>>(Q,  ws + QBF_OFF, Bc * Sc * Dc);
    cvt_bf16_kernel<<<(Dc * Dc) / 8 / 256, 256, 0, stream>>>(Wq, ws + WBF_OFF,               Dc * Dc);
    cvt_bf16_kernel<<<(Dc * Dc) / 8 / 256, 256, 0, stream>>>(Wk, ws + WBF_OFF + Dc * Dc,     Dc * Dc);
    cvt_bf16_kernel<<<(Dc * Dc) / 8 / 256, 256, 0, stream>>>(Wv, ws + WBF_OFF + 2 * Dc * Dc, Dc * Dc);

    // 1) QKV projection: 49152 wave-jobs / 8 waves per block
    qkv_proj_kernel<<<6144, 256, 0, stream>>>(ws, bq, bk, bv, ws);

    // 2) attention: B*H * (S/32) = 8192 blocks
    attn_kernel<<<8192, 256, 0, stream>>>(ws, length, ctx, attn);
}
